// RelationAwareAttention_17334488006785
// MI455X (gfx1250) — matrix operand order repaired and checked by reading, not compiled, not run on hardware
//
#include <hip/hip_runtime.h>
#include <math.h>

// ---------------------------------------------------------------------------
// Relation-aware attention, MI455X (gfx1250), wave32 + v_wmma_f32_16x16x32_bf16.
//   B=4 H=8 L=1024 D=64 P=512.
//   Pass 0: one-time fp32->bf16 conversion of Q/K/rk (row-major) and V/rv
//           (transposed, D-major) into d_ws -> every WMMA fragment becomes two
//           contiguous 16B loads (global_load_b128 / ds_load_b128), no per-
//           fragment conversion VALU.
//   Pass 1: one workgroup per (b,h, 16-row stripe): S and qr stripes in LDS,
//           exact softmax, p_attn streamed to HBM once, P*V + w*rv on WMMA.
// ---------------------------------------------------------------------------

typedef __attribute__((ext_vector_type(16))) __bf16       v16bf;
typedef __attribute__((ext_vector_type(8)))  float        v8f;
typedef __attribute__((ext_vector_type(4)))  unsigned int v4u;

#define BB 4
#define HH 8
#define LLEN 1024
#define DD 64
#define PPATH 512
#define ROWS 16
#define NRB (LLEN / ROWS)   // 64 row stripes
#define THREADS 128
#define NWAVES 4

// padded LDS strides (elements); all row strides multiple of 16 bytes -> 16B-aligned chunks
#define SLDF 1028   // f32 score stripe
#define QLDF 516    // f32 qr / path-weight stripe
#define SLDB 1040   // bf16 prob stripe
#define WLDB 528    // bf16 path-weight stripe
#define QLDQ 80     // bf16 staged Q stripe

__device__ __forceinline__ unsigned short bfbits(float f) {
  unsigned u = __builtin_bit_cast(unsigned, f);
  return (unsigned short)((u + 0x7fffu + ((u >> 16) & 1u)) >> 16);  // RNE
}
__device__ __forceinline__ __bf16 f2bf(float f) {
  unsigned short s = bfbits(f);
  return __builtin_bit_cast(__bf16, s);
}
__device__ __forceinline__ unsigned pack2(float a, float b) {
  return (unsigned)bfbits(a) | ((unsigned)bfbits(b) << 16);
}

union FragU { v4u u[2]; v16bf b; };

// A tile 16x32 (MxK) from bf16 memory, row stride ld (elements), K-contiguous.
// CDNA5 layout: lane half 0 -> K {0..7,16..23}, half 1 -> +8. Two 16B chunks.
__device__ __forceinline__ v16bf fragA_bf(const __bf16* src, int ld) {
  const int lane = threadIdx.x & 31;
  const int m = lane & 15, half = lane >> 4;
  const __bf16* p = src + (size_t)m * ld + half * 8;
  FragU f;
  f.u[0] = *(const v4u*)p;          // K base+0..7
  f.u[1] = *(const v4u*)(p + 16);   // K base+16..23
  return f.b;
}

// B tile 32x16 (KxN) with B[k][n] = W[(n0+n)*ldw + kc + k] (K contiguous in W).
// Lane half 0 -> K {0..7,16..23}, half 1 -> +8 (matches fragA): two 16B chunks.
__device__ __forceinline__ v16bf fragB_bf(const __bf16* W, int ldw, int n0, int kc) {
  const int lane = threadIdx.x & 31;
  const int n = lane & 15, half = lane >> 4;
  const __bf16* p = W + (size_t)(n0 + n) * ldw + kc + half * 8;
  FragU f;
  f.u[0] = *(const v4u*)p;
  f.u[1] = *(const v4u*)(p + 16);
  return f.b;
}

__device__ __forceinline__ v8f wmma_bf16(v16bf a, v16bf b, v8f c) {
  return __builtin_amdgcn_wmma_f32_16x16x32_bf16(false, a, false, b,
                                                 (short)0, c, false, false);
}

// ---- pass-0 kernels ------------------------------------------------------

// elementwise fp32 -> bf16 (n multiple of 4)
__global__ void cvt_bf16_kernel(const float* __restrict__ in,
                                __bf16* __restrict__ out, int n4) {
  int i = blockIdx.x * blockDim.x + threadIdx.x;
  if (i >= n4) return;
  float4 v = ((const float4*)in)[i];
  uint2 o;
  o.x = pack2(v.x, v.y);
  o.y = pack2(v.z, v.w);
  ((uint2*)out)[i] = o;
}

// per-(b,h) transpose + convert: in (BH,R,C) f32 -> out (BH,C,R) bf16
__global__ void transpose_cvt_kernel(const float* __restrict__ in,
                                     __bf16* __restrict__ out, int R, int C) {
  __shared__ float tile[32][33];
  const int bh = blockIdx.z;
  const float* src = in + (size_t)bh * R * C;
  __bf16* dst = out + (size_t)bh * R * C;
  const int r0 = blockIdx.x * 32, c0 = blockIdx.y * 32;
  for (int i = threadIdx.y; i < 32; i += 8)
    tile[i][threadIdx.x] = src[(size_t)(r0 + i) * C + c0 + threadIdx.x];
  __syncthreads();
  for (int i = threadIdx.y; i < 32; i += 8)
    dst[(size_t)(c0 + i) * R + r0 + threadIdx.x] = f2bf(tile[threadIdx.x][i]);
}

// ---- main fused kernel ---------------------------------------------------

__global__ void ra_attn_kernel(const __bf16* __restrict__ Qbf,
                               const __bf16* __restrict__ Kbf,
                               const __bf16* __restrict__ VT,
                               const __bf16* __restrict__ RKbf,
                               const __bf16* __restrict__ RVT,
                               const int*    __restrict__ PM,
                               const int*    __restrict__ MSK,
                               float* __restrict__ OutAttn,
                               float* __restrict__ OutP) {
  extern __shared__ char smem_raw[];
  float*  Sf  = (float*)smem_raw;               // ROWS x SLDF (scores f32)
  float*  QRf = Sf + ROWS * SLDF;               // ROWS x QLDF (qr -> w, f32)
  __bf16* Sbf = (__bf16*)(QRf + ROWS * QLDF);   // ROWS x SLDB (probs bf16)
  __bf16* Wbf = Sbf + ROWS * SLDB;              // ROWS x WLDB (w bf16)
  __bf16* Qsb = Wbf + ROWS * WLDB;              // ROWS x QLDQ (Q stripe bf16)

  const int bid = blockIdx.x;
  const int bh = bid / NRB;
  const int rb = bid % NRB;
  const int b  = bh / HH;
  const int i0 = rb * ROWS;

  const __bf16* q  = Qbf  + (size_t)bh * LLEN * DD + (size_t)i0 * DD;
  const __bf16* k  = Kbf  + (size_t)bh * LLEN * DD;
  const __bf16* vt = VT   + (size_t)bh * LLEN * DD;   // (D x L)
  const __bf16* rk = RKbf + (size_t)bh * PPATH * DD;
  const __bf16* rvt= RVT  + (size_t)bh * PPATH * DD;  // (D x P)
  const int* pm  = PM  + (size_t)b * LLEN * LLEN + (size_t)i0 * LLEN;
  const int* msk = MSK + (size_t)b * LLEN * LLEN + (size_t)i0 * LLEN;

  const int tid  = threadIdx.x;
  const int lane = tid & 31;
  const int wave = tid >> 5;
  const int half = lane >> 4;
  const int nn   = lane & 15;

  // ---- stage Q stripe (bf16, padded stride) ----
  for (int idx = tid; idx < ROWS * (DD / 2); idx += THREADS) {
    const int r = idx >> 5, c = idx & 31;       // dword granules (2 bf16)
    ((unsigned*)(Qsb))[r * (QLDQ / 2) + c] = ((const unsigned*)q)[r * (DD / 2) + c];
  }
  __syncthreads();

  const v16bf a0 = fragA_bf(Qsb, QLDQ);         // K 0..31
  const v16bf a1 = fragA_bf(Qsb + 32, QLDQ);    // K 32..63

  // ---- qr = Qblk * rk^T  (16 x 512): 8 col tiles per wave ----
  for (int ct = wave; ct < PPATH / 16; ct += NWAVES) {
    const int p0 = ct * 16;
    v8f c = {};
    c = wmma_bf16(a0, fragB_bf(rk, DD, p0, 0),  c);
    c = wmma_bf16(a1, fragB_bf(rk, DD, p0, 32), c);
#pragma unroll
    for (int r = 0; r < 8; ++r)
      QRf[(r + 8 * half) * QLDF + p0 + nn] = c[r];
  }

  // ---- S = Qblk * K^T  (16 x 1024): 16 col tiles per wave ----
  for (int ct = wave; ct < LLEN / 16; ct += NWAVES) {
    const int j0 = ct * 16;
    v8f c = {};
    c = wmma_bf16(a0, fragB_bf(k, DD, j0, 0),  c);
    c = wmma_bf16(a1, fragB_bf(k, DD, j0, 32), c);
#pragma unroll
    for (int r = 0; r < 8; ++r)
      Sf[(r + 8 * half) * SLDF + j0 + nn] = c[r];
  }
  __syncthreads();

  // ---- relation gather + mask + exact softmax; rows are wave-private ----
  const float scale = 0.125f;                   // 1/sqrt(64)
  const int r0 = wave * (ROWS / NWAVES);        // 4 rows per wave
  const int jb = lane * (LLEN / 32);            // lane owns 32 contiguous cols
  for (int r = r0; r < r0 + ROWS / NWAVES; ++r) {
    float* srow = Sf + r * SLDF;
    float* qrow = QRf + r * QLDF;               // qr, later reused as w
    const int* pmrow = pm + (size_t)r * LLEN;
    const int* mrow  = msk + (size_t)r * LLEN;

    float mx = -3.0e38f;
#pragma unroll
    for (int t = 0; t < 8; ++t) {
      const int j = jb + t * 4;
      float4 s4 = *(float4*)(srow + j);
      const int4 p4 = *(const int4*)(pmrow + j);
      const int4 m4 = *(const int4*)(mrow + j);
      float v0 = (s4.x + (((unsigned)p4.x < PPATH) ? qrow[p4.x] : 0.0f)) * scale;
      float v1 = (s4.y + (((unsigned)p4.y < PPATH) ? qrow[p4.y] : 0.0f)) * scale;
      float v2 = (s4.z + (((unsigned)p4.z < PPATH) ? qrow[p4.z] : 0.0f)) * scale;
      float v3 = (s4.w + (((unsigned)p4.w < PPATH) ? qrow[p4.w] : 0.0f)) * scale;
      if (m4.x == 0) v0 = -1.0e9f;
      if (m4.y == 0) v1 = -1.0e9f;
      if (m4.z == 0) v2 = -1.0e9f;
      if (m4.w == 0) v3 = -1.0e9f;
      s4 = make_float4(v0, v1, v2, v3);
      *(float4*)(srow + j) = s4;
      mx = fmaxf(mx, fmaxf(fmaxf(v0, v1), fmaxf(v2, v3)));
    }
#pragma unroll
    for (int off = 16; off > 0; off >>= 1) mx = fmaxf(mx, __shfl_xor(mx, off, 32));

    float sum = 0.0f;
#pragma unroll
    for (int t = 0; t < 8; ++t) {
      const int j = jb + t * 4;
      float4 s4 = *(float4*)(srow + j);
      s4.x = __expf(s4.x - mx); s4.y = __expf(s4.y - mx);
      s4.z = __expf(s4.z - mx); s4.w = __expf(s4.w - mx);
      *(float4*)(srow + j) = s4;
      sum += (s4.x + s4.y) + (s4.z + s4.w);
    }
#pragma unroll
    for (int off = 16; off > 0; off >>= 1) sum += __shfl_xor(sum, off, 32);
    const float inv = 1.0f / sum;

    // retire qr row: becomes path-weight accumulator w[r][*]
#pragma unroll
    for (int t = 0; t < 4; ++t)
      *(float4*)(qrow + lane * 16 + t * 4) = make_float4(0.f, 0.f, 0.f, 0.f);

    float* prow = OutP + ((size_t)bh * LLEN + i0 + r) * LLEN;
#pragma unroll
    for (int t = 0; t < 8; ++t) {
      const int j = jb + t * 4;
      float4 s4 = *(float4*)(srow + j);
      s4.x *= inv; s4.y *= inv; s4.z *= inv; s4.w *= inv;
      *(float4*)(prow + j) = s4;                            // p_attn out (b128)
      uint2 pk; pk.x = pack2(s4.x, s4.y); pk.y = pack2(s4.z, s4.w);
      *(uint2*)(Sbf + r * SLDB + j) = pk;                   // bf16 probs for P*V
      const int4 p4 = *(const int4*)(pmrow + j);
      if ((unsigned)p4.x < PPATH) atomicAdd(&qrow[p4.x], s4.x);
      if ((unsigned)p4.y < PPATH) atomicAdd(&qrow[p4.y], s4.y);
      if ((unsigned)p4.z < PPATH) atomicAdd(&qrow[p4.z], s4.z);
      if ((unsigned)p4.w < PPATH) atomicAdd(&qrow[p4.w], s4.w);
    }
  }
  // convert this wave's path-weight rows to bf16 (wave-private, no barrier yet)
  for (int r = r0; r < r0 + ROWS / NWAVES; ++r) {
    const float* qrow = QRf + r * QLDF;
#pragma unroll
    for (int t = 0; t < 4; ++t) {
      const int p = lane * 16 + t * 4;
      float4 w4 = *(const float4*)(qrow + p);
      uint2 pk; pk.x = pack2(w4.x, w4.y); pk.y = pack2(w4.z, w4.w);
      *(uint2*)(Wbf + r * WLDB + p) = pk;
    }
  }
  __syncthreads();

  // ---- Out(16x64) = P(16x1024)*V + w(16x512)*rv : one 16x16 tile per wave ----
  const int n0 = wave * 16;
  v8f acc = {};
  for (int j0 = 0; j0 < LLEN; j0 += 32)
    acc = wmma_bf16(fragA_bf(Sbf + j0, SLDB), fragB_bf(vt, LLEN, n0, j0), acc);
  for (int p0 = 0; p0 < PPATH; p0 += 32)
    acc = wmma_bf16(fragA_bf(Wbf + p0, WLDB), fragB_bf(rvt, PPATH, n0, p0), acc);

  float* orow = OutAttn + ((size_t)bh * LLEN + i0) * DD;
#pragma unroll
  for (int r = 0; r < 8; ++r)
    orow[(size_t)(r + 8 * half) * DD + n0 + nn] = acc[r];
}

// ---- host launcher -------------------------------------------------------

extern "C" void kernel_launch(void* const* d_in, const int* in_sizes, int n_in,
                              void* d_out, int out_size, void* d_ws, size_t ws_size,
                              hipStream_t stream) {
  const float* Q  = (const float*)d_in[0];
  const float* K  = (const float*)d_in[1];
  const float* V  = (const float*)d_in[2];
  const float* RK = (const float*)d_in[3];
  const float* RV = (const float*)d_in[4];
  const int*   PM = (const int*)d_in[5];
  const int*   MS = (const int*)d_in[6];
  float* outA = (float*)d_out;
  float* outP = outA + (size_t)BB * HH * LLEN * DD;

  const size_t N1 = (size_t)BB * HH * LLEN * DD;    // 2,097,152
  const size_t N2 = (size_t)BB * HH * PPATH * DD;   // 1,048,576
  __bf16* qbf  = (__bf16*)d_ws;
  __bf16* kbf  = qbf + N1;
  __bf16* vt   = kbf + N1;
  __bf16* rkbf = vt + N1;
  __bf16* rvt  = rkbf + N2;                          // total 16 MB of d_ws

  // pass 0: one-time bf16 conversion / transposition
  cvt_bf16_kernel<<<dim3((unsigned)(N1 / 4 / 256)), 256, 0, stream>>>(Q, qbf, (int)(N1 / 4));
  cvt_bf16_kernel<<<dim3((unsigned)(N1 / 4 / 256)), 256, 0, stream>>>(K, kbf, (int)(N1 / 4));
  cvt_bf16_kernel<<<dim3((unsigned)(N2 / 4 / 256)), 256, 0, stream>>>(RK, rkbf, (int)(N2 / 4));
  transpose_cvt_kernel<<<dim3(LLEN / 32, DD / 32, BB * HH), dim3(32, 8), 0, stream>>>(V, vt, LLEN, DD);
  transpose_cvt_kernel<<<dim3(PPATH / 32, DD / 32, BB * HH), dim3(32, 8), 0, stream>>>(RV, rvt, PPATH, DD);

  // pass 1: fused attention
  const size_t shbytes = (size_t)(ROWS * SLDF + ROWS * QLDF) * sizeof(float) +
                         (size_t)(ROWS * SLDB + ROWS * WLDB + ROWS * QLDQ) * sizeof(__bf16);
  hipFuncSetAttribute((const void*)ra_attn_kernel,
                      hipFuncAttributeMaxDynamicSharedMemorySize, (int)shbytes);
  ra_attn_kernel<<<dim3(BB * HH * NRB), THREADS, shbytes, stream>>>(
      qbf, kbf, vt, rkbf, rvt, PM, MS, outA, outP);
}
